// GraphHeart_36996848288031
// MI455X (gfx1250) — matrix operand order repaired and checked by reading, not compiled, not run on hardware
//
#include <hip/hip_runtime.h>
#include <hip/hip_bf16.h>

// ============================================================================
// ST-GCN VAE forward for MI455X (gfx1250, wave32, WMMA).
// Dense contractions -> v_wmma_f32_16x16x32_f16 (f16 in, f32 accumulate),
// operands staged through LDS (vectorized b128 loads/stores on interior
// tiles, ds_load_b128 fragment reads).
// Spline conv -> per-edge effective weight precompute + atomic scatter.
// ============================================================================

typedef __attribute__((ext_vector_type(16))) _Float16 v16h;
typedef __attribute__((ext_vector_type(8)))  _Float16 v8h;
typedef __attribute__((ext_vector_type(8)))  float    v8f;

#define BATCH 2

__device__ __forceinline__ float elu1(float x) {
  return x > 0.f ? x : (__expf(x) - 1.f);
}

// ---------------------------------------------------------------------------
// WMMA GEMM:  C[b][M,N] = act( A[b][M,K] @ B[b][K,N] + bias )
//   A row-major (lda), C row-major (ldc)
//   transB==0: B is KxN row-major (ldb); transB==1: B is NxK row-major (B^T)
//   biasMode: 0 none, 1 per-row bias[M], 2 per-col bias[N];  act: 0 none, 1 ELU
//
// 128 threads = 4 waves; block tile 32x32, one 16x16 WMMA tile per wave.
// Tiles staged fp32->f16 into LDS; A stored [M][K], B stored transposed [N][K]
// so every lane's fragment is two contiguous 16B LDS reads (ISA 7.12.2):
//   A frag (lane m,g): K in [8g..8g+7] and [16+8g..16+8g+7] of row m
//   B frag (lane n,g): K in [8g..8g+7] and [16+8g..16+8g+7] of row n
// Rows padded to 40 halves (20 banks) -> conflict-free b128 reads.
// Interior tiles use a block-uniform fast path: 2x float4 global loads +
// one v8h (16B) ds store per thread; clamp+select only on edge tiles.
// ---------------------------------------------------------------------------
__global__ __launch_bounds__(128)
void gemm_wmma_f16(const float* __restrict__ A, const float* __restrict__ Bm,
                   float* __restrict__ C, const float* __restrict__ bias,
                   int M, int N, int K, int lda, int ldb, int ldc,
                   long sA, long sB, long sC,
                   int transB, int biasMode, int act)
{
  __shared__ _Float16 As[32][40];
  __shared__ _Float16 Bs[32][40];   // stored as [n][k]

  const int batch = blockIdx.z;
  const float* Ab = A + (long)batch * sA;
  const float* Bb = Bm + (long)batch * sB;
  float* Cb = C + (long)batch * sC;

  const int t    = threadIdx.x;
  const int wave = t >> 5;
  const int lane = t & 31;
  const int m16  = lane & 15;
  const int g    = lane >> 4;
  const int blkM = blockIdx.y * 32;
  const int blkN = blockIdx.x * 32;
  const int am   = (wave >> 1) * 16 + m16;  // local A row for this lane
  const int bn   = (wave & 1) * 16 + m16;   // local B col (row of Bs)
  const int tileM = blkM + (wave >> 1) * 16;
  const int tileN = blkN + (wave & 1) * 16;
  const int col  = tileN + m16;

  // staging coordinates: thread owns 8 consecutive elements of one row
  const int tr  = t >> 2;          // 0..31
  const int tc8 = (t & 3) * 8;     // 0,8,16,24

  const bool fullM = (blkM + 32 <= M);
  const bool fullN = (blkN + 32 <= N);
  const bool vecA  = ((lda & 3) == 0);
  const bool vecB  = ((ldb & 3) == 0);

  v8f acc = {};
  for (int k0 = 0; k0 < K; k0 += 32) {
    const bool fullK = (k0 + 32 <= K);
    __syncthreads();

    // ---- stage A tile ----
    if (fullM && fullK) {                 // block-uniform fast path
      const float* src = Ab + (long)(blkM + tr) * lda + (k0 + tc8);
      if (k0 + 32 < K) __builtin_prefetch(src + 32, 0, 1);
      float vv[8];
      if (vecA) {
        const float4 p0 = ((const float4*)src)[0];
        const float4 p1 = ((const float4*)src)[1];
        vv[0] = p0.x; vv[1] = p0.y; vv[2] = p0.z; vv[3] = p0.w;
        vv[4] = p1.x; vv[5] = p1.y; vv[6] = p1.z; vv[7] = p1.w;
      } else {
#pragma unroll
        for (int j = 0; j < 8; ++j) vv[j] = src[j];
      }
      v8h pk;
#pragma unroll
      for (int j = 0; j < 8; ++j) pk[j] = (_Float16)vv[j];
      *(v8h*)&As[tr][tc8] = pk;           // single 16B ds_store
    } else {                              // edge tile: clamp + select
      const int ar  = blkM + tr;
      const int arc = (ar < M - 1) ? ar : (M - 1);
      const float* Arow = Ab + (long)arc * lda;
      v8h pk;
#pragma unroll
      for (int j = 0; j < 8; ++j) {
        const int k  = k0 + tc8 + j;
        const int kc = (k < K - 1) ? k : (K - 1);
        const float v = Arow[kc];
        pk[j] = (_Float16)((ar < M && k < K) ? v : 0.f);
      }
      *(v8h*)&As[tr][tc8] = pk;
    }

    // ---- stage B tile into Bs[n][k] ----
    if (!transB) {
      if (fullN && fullK) {
        const int k = k0 + tr;
        const float* src = Bb + (long)k * ldb + (blkN + tc8);
        if (k0 + 32 < K) __builtin_prefetch(src + 32L * ldb, 0, 1);
        float vv[8];
        if (vecB) {
          const float4 p0 = ((const float4*)src)[0];
          const float4 p1 = ((const float4*)src)[1];
          vv[0] = p0.x; vv[1] = p0.y; vv[2] = p0.z; vv[3] = p0.w;
          vv[4] = p1.x; vv[5] = p1.y; vv[6] = p1.z; vv[7] = p1.w;
        } else {
#pragma unroll
          for (int j = 0; j < 8; ++j) vv[j] = src[j];
        }
#pragma unroll
        for (int j = 0; j < 8; ++j)       // LDS transpose (scalar stores)
          Bs[tc8 + j][tr] = (_Float16)vv[j];
      } else {
        const int k  = k0 + tr;
        const int kc = (k < K - 1) ? k : (K - 1);
        const float* Brow = Bb + (long)kc * ldb;
#pragma unroll
        for (int j = 0; j < 8; ++j) {
          const int n  = blkN + tc8 + j;
          const int nc = (n < N - 1) ? n : (N - 1);
          const float v = Brow[nc];
          Bs[tc8 + j][tr] = (_Float16)((k < K && n < N) ? v : 0.f);
        }
      }
    } else {
      if (fullN && fullK) {
        const float* src = Bb + (long)(blkN + tr) * ldb + (k0 + tc8);
        if (k0 + 32 < K) __builtin_prefetch(src + 32, 0, 1);
        float vv[8];
        if (vecB) {
          const float4 p0 = ((const float4*)src)[0];
          const float4 p1 = ((const float4*)src)[1];
          vv[0] = p0.x; vv[1] = p0.y; vv[2] = p0.z; vv[3] = p0.w;
          vv[4] = p1.x; vv[5] = p1.y; vv[6] = p1.z; vv[7] = p1.w;
        } else {
#pragma unroll
          for (int j = 0; j < 8; ++j) vv[j] = src[j];
        }
        v8h pk;
#pragma unroll
        for (int j = 0; j < 8; ++j) pk[j] = (_Float16)vv[j];
        *(v8h*)&Bs[tr][tc8] = pk;
      } else {
        const int n  = blkN + tr;
        const int nc = (n < N - 1) ? n : (N - 1);
        const float* Brow = Bb + (long)nc * ldb;
        v8h pk;
#pragma unroll
        for (int j = 0; j < 8; ++j) {
          const int k  = k0 + tc8 + j;
          const int kc = (k < K - 1) ? k : (K - 1);
          const float v = Brow[kc];
          pk[j] = (_Float16)((k < K && n < N) ? v : 0.f);
        }
        *(v8h*)&Bs[tr][tc8] = pk;
      }
    }
    __syncthreads();

    // ---- fragments: two contiguous 16B LDS reads each ----
    const v8h alo = *(const v8h*)&As[am][8 * g];
    const v8h ahi = *(const v8h*)&As[am][16 + 8 * g];
    const v8h blo = *(const v8h*)&Bs[bn][8 * g];
    const v8h bhi = *(const v8h*)&Bs[bn][16 + 8 * g];
    const v16h a = __builtin_shufflevector(alo, ahi, 0, 1, 2, 3, 4, 5, 6, 7,
                                           8, 9, 10, 11, 12, 13, 14, 15);
    const v16h b = __builtin_shufflevector(blo, bhi, 0, 1, 2, 3, 4, 5, 6, 7,
                                           8, 9, 10, 11, 12, 13, 14, 15);
    acc = __builtin_amdgcn_wmma_f32_16x16x32_f16(false, a, false, b,
                                                 (short)0, acc, false, false);
  }

#pragma unroll
  for (int i = 0; i < 8; ++i) {
    const int r = tileM + i + 8 * g;
    if (r < M && col < N) {
      float v = acc[i];
      if (biasMode == 1) v += bias[r];
      else if (biasMode == 2) v += bias[col];
      if (act) v = elu1(v);
      Cb[(long)r * ldc + col] = v;
    }
  }
}

// ---------------------------------------------------------------------------
// x [B,V,C,T] -> xg [(t*B+n), V, C]   (gather for spline conv over graph)
// ---------------------------------------------------------------------------
__global__ void k_to_xg(const float* __restrict__ x, float* __restrict__ xg,
                        int Vn, int Cc, int T, long total)
{
  long i = (long)blockIdx.x * blockDim.x + threadIdx.x;
  if (i >= total) return;
  int t = (int)(i % T); long r = i / T;
  int c = (int)(r % Cc); r /= Cc;
  int v = (int)(r % Vn); int n = (int)(r / Vn);
  xg[(((long)t * BATCH + n) * Vn + v) * Cc + c] = x[i];
}

// ---------------------------------------------------------------------------
// Effective per-edge spline weight: Weff[e] = sum_bits basis(e,bits)*Ws[idx]
// DIM=3, K1=3 -> 8 corner terms; Ws is [27, cin, cout].
// ---------------------------------------------------------------------------
__global__ void k_edge_weights(const float* __restrict__ ea,
                               const float* __restrict__ Ws,
                               float* __restrict__ weff,
                               int E, int cin, int cout, long total)
{
  long i = (long)blockIdx.x * blockDim.x + threadIdx.x;
  if (i >= total) return;
  int d = (int)(i % cout); long r = i / cout;
  int c = (int)(r % cin);  int e = (int)(r / cin);

  float f[3]; int lo[3];
#pragma unroll
  for (int k = 0; k < 3; ++k) {
    float p = fminf(fmaxf(ea[(long)e * 3 + k], 0.f), 1.f) * 2.f;
    float fl = fminf(fmaxf(floorf(p), 0.f), 1.f);
    lo[k] = (int)fl;
    f[k]  = p - fl;
  }
  float acc = 0.f;
#pragma unroll
  for (int bits = 0; bits < 8; ++bits) {
    float basis = 1.f; int idx = 0; int stride = 1;
#pragma unroll
    for (int k = 0; k < 3; ++k) {
      int bv = (bits >> k) & 1;
      basis *= bv ? f[k] : (1.f - f[k]);
      idx += (lo[k] + bv) * stride;
      stride *= 3;
    }
    acc += basis * Ws[((long)idx * cin + c) * cout + d];
  }
  weff[i] = acc;
}

// ---------------------------------------------------------------------------
// Scatter messages: h[g, dst[e], d] += xg[g, src[e], :] . Weff[e, :, d]
// ---------------------------------------------------------------------------
__global__ void k_spline_scatter(const float* __restrict__ xg,
                                 const float* __restrict__ weff,
                                 const int* __restrict__ ei,
                                 float* __restrict__ h,
                                 int Vn, int E, int cin, int cout, long total)
{
  long i = (long)blockIdx.x * blockDim.x + threadIdx.x;
  if (i >= total) return;
  int d = (int)(i % cout); long r = i / cout;
  int e = (int)(r % E);    int g = (int)(r / E);
  int src = ei[e];
  int dst = ei[E + e];
  const float* xv = xg + ((long)g * Vn + src) * cin;
  const float* w  = weff + (long)e * cin * cout + d;
  float acc = 0.f;
  for (int c = 0; c < cin; ++c) acc += xv[c] * w[(long)c * cout];
  atomicAdd(&h[((long)g * Vn + dst) * cout + d], acc);
}

// h = elu( elu(h_spline) + res )   (res already ELU'd by GEMM epilogue)
__global__ void k_fuse(float* __restrict__ h, const float* __restrict__ res,
                       long total)
{
  long i = (long)blockIdx.x * blockDim.x + threadIdx.x;
  if (i >= total) return;
  h[i] = elu1(elu1(h[i]) + res[i]);
}

// tt[S, J] (J=(n*V+v)*D+d) -> x_out[((n*V+v)*D+d)*S + s] = elu(tt + bt[s])
__global__ void k_temporal_out(const float* __restrict__ tt,
                               const float* __restrict__ bt,
                               float* __restrict__ xo,
                               int S, long J, long total)
{
  long i = (long)blockIdx.x * blockDim.x + threadIdx.x;
  if (i >= total) return;
  int s = (int)(i % S);
  long j = i / S;
  xo[i] = elu1(tt[(long)s * J + j] + bt[s]);
}

// x [B, V, C, T] -> y [B, C, V, T]
__global__ void k_tr_vc(const float* __restrict__ x, float* __restrict__ y,
                        int Vn, int Cc, int T, long total)
{
  long i = (long)blockIdx.x * blockDim.x + threadIdx.x;
  if (i >= total) return;
  int t = (int)(i % T); long r = i / T;
  int c = (int)(r % Cc); r /= Cc;
  int v = (int)(r % Vn); int n = (int)(r / Vn);
  y[(((long)n * Cc + c) * Vn + v) * T + t] = x[i];
}

// z = mu + eps * exp(0.5*logvar)
__global__ void k_reparam(const float* __restrict__ mu,
                          const float* __restrict__ logvar,
                          const float* __restrict__ eps,
                          float* __restrict__ z, long total)
{
  long i = (long)blockIdx.x * blockDim.x + threadIdx.x;
  if (i >= total) return;
  z[i] = mu[i] + eps[i] * __expf(0.5f * logvar[i]);
}

// ============================ host-side drivers ============================

static inline unsigned cdiv(long a, long b) { return (unsigned)((a + b - 1) / b); }

static void gemm(hipStream_t s, const float* A, const float* B, float* C,
                 const float* bias, int M, int N, int K, int lda, int ldb,
                 int ldc, long sA, long sB, long sC, int batches, int transB,
                 int biasMode, int act)
{
  dim3 grid(cdiv(N, 32), cdiv(M, 32), batches);
  gemm_wmma_f16<<<grid, 128, 0, s>>>(A, B, C, bias, M, N, K, lda, ldb, ldc,
                                     sA, sB, sC, transB, biasMode, act);
}

struct BlockP { const float *Ws, *Wr, *bg, *Wres, *bres, *Wt, *bt; };

static void run_block(hipStream_t s, const float* x, float* x_out,
                      int Vn, int cin, int cout, int Tin, int Tout,
                      const int* ei, const float* ea, int E, const BlockP& p,
                      float* xg, float* h, float* res, float* weff, float* tt)
{
  const int G = Tin * BATCH;
  const long nx = (long)BATCH * Vn * cin * Tin;
  k_to_xg<<<cdiv(nx, 256), 256, 0, s>>>(x, xg, Vn, cin, Tin, nx);

  const long MGV = (long)G * Vn;
  // h = xg @ Wr + bg  (per-column bias, no activation yet)
  gemm(s, xg, p.Wr, h, p.bg, (int)MGV, cout, cin, cin, cout, cout,
       0, 0, 0, 1, 0, 2, 0);
  // res = elu(xg @ Wres^T + bres)
  gemm(s, xg, p.Wres, res, p.bres, (int)MGV, cout, cin, cin, cin, cout,
       0, 0, 0, 1, 1, 2, 1);

  const long nEw = (long)E * cin * cout;
  k_edge_weights<<<cdiv(nEw, 256), 256, 0, s>>>(ea, p.Ws, weff, E, cin, cout, nEw);

  const long nSc = (long)G * E * cout;
  k_spline_scatter<<<cdiv(nSc, 256), 256, 0, s>>>(xg, weff, ei, h, Vn, E,
                                                  cin, cout, nSc);
  const long nH = MGV * cout;
  k_fuse<<<cdiv(nH, 256), 256, 0, s>>>(h, res, nH);

  // temporal: tt[S, J] = Wt[S,Tin] @ h[Tin, J],  J = B*Vn*cout
  const long J = (long)BATCH * Vn * cout;
  gemm(s, p.Wt, h, tt, nullptr, Tout, (int)J, Tin, Tin, (int)J, (int)J,
       0, 0, 0, 1, 0, 0, 0);
  const long nO = J * Tout;
  k_temporal_out<<<cdiv(nO, 256), 256, 0, s>>>(tt, p.bt, x_out, Tout, J, nO);
}

static void run_pool(hipStream_t s, const float* P, const float* x, float* y,
                     int Vout, int Vin, int N)
{
  gemm(s, P, x, y, nullptr, Vout, N, Vin, Vin, N, N,
       0, (long)Vin * N, (long)Vout * N, BATCH, 0, 0, 0);
}

static void run_c1x1(hipStream_t s, const float* W, const float* b,
                     const float* x, float* y, int dout, int cin, int act)
{
  const int N = 128 * 8;  // V4 * NS4
  gemm(s, W, x, y, b, dout, N, cin, cin, N, N,
       0, (long)cin * N, (long)dout * N, BATCH, 0, 1, act);
}

// ================================ entry ====================================

extern "C" void kernel_launch(void* const* d_in, const int* in_sizes, int n_in,
                              void* d_out, int out_size, void* d_ws,
                              size_t ws_size, hipStream_t stream)
{
  (void)in_sizes; (void)n_in; (void)out_size; (void)ws_size;

  const float* data = (const float*)d_in[0];
  const float* eps  = (const float*)d_in[1];
  const int*   ei[4]; const float* ea[4];
  for (int l = 0; l < 4; ++l) {
    ei[l] = (const int*)d_in[2 + 2 * l];
    ea[l] = (const float*)d_in[3 + 2 * l];
  }
  const float* P01 = (const float*)d_in[10];
  const float* P12 = (const float*)d_in[11];
  const float* P23 = (const float*)d_in[12];
  const float* P34 = (const float*)d_in[13];
  const float* P43 = (const float*)d_in[14];
  const float* P32 = (const float*)d_in[15];
  const float* P21 = (const float*)d_in[16];
  const float* P10 = (const float*)d_in[17];

  BlockP blk[8];  // c1 c2 c3 c4 d4 d3 d2 d1
  for (int b = 0; b < 8; ++b) {
    const int base = 18 + 7 * b;
    blk[b].Ws   = (const float*)d_in[base + 0];
    blk[b].Wr   = (const float*)d_in[base + 1];
    blk[b].bg   = (const float*)d_in[base + 2];
    blk[b].Wres = (const float*)d_in[base + 3];
    blk[b].bres = (const float*)d_in[base + 4];
    blk[b].Wt   = (const float*)d_in[base + 5];
    blk[b].bt   = (const float*)d_in[base + 6];
  }
  const float* fce1W  = (const float*)d_in[74];
  const float* fce1b  = (const float*)d_in[75];
  const float* fce21W = (const float*)d_in[76];
  const float* fce21b = (const float*)d_in[77];
  const float* fce22W = (const float*)d_in[78];
  const float* fce22b = (const float*)d_in[79];
  const float* fcd3W  = (const float*)d_in[80];
  const float* fcd3b  = (const float*)d_in[81];
  const float* fcd4W  = (const float*)d_in[82];
  const float* fcd4b  = (const float*)d_in[83];

  // -------- workspace carve-out (floats); peak ~19.46M floats (~78 MB) -----
  float* ws = (float*)d_ws;
  float* XA = ws;                 // 2,000,000  (activations ping)
  float* XB = XA + 2000000;       // 2,000,000  (activations pong)
  float* XG = XB + 2000000;       // 2,000,000  (spline-gather layout)
  float* H  = XG + 2000000;       // 3,300,000  (spline out)
  float* RS = H  + 3300000;       // 3,300,000  (residual)
  float* TT = RS + 3300000;       // 2,000,000  (temporal GEMM out)
  float* WE = TT + 2000000;       // 4,200,000  (per-edge effective weights)
  float* M1 = WE + 4200000;       //   262,144  (mid buffer)
  float* M2 = M1 + 262144;        //   262,144  (mid buffer)
  float* ZB = M2 + 262144;        //   131,072  (latent z)

  float* recon  = (float*)d_out;           // (B*2048, 1, 100) = 409600
  float* mu     = recon + 409600;          // (2,64,128,8)     = 131072
  float* logvar = mu + 131072;             // (2,64,128,8)     = 131072

  // ---------------- encoder ----------------
  run_block(stream, data, XA, 2048, 1, 8, 100, 60, ei[0], ea[0], 16384,
            blk[0], XG, H, RS, WE, TT);                     // c1 -> (2,2048,8,60)
  run_pool(stream, P01, XA, XB, 1024, 2048, 8 * 60);        // -> (2,1024,8,60)
  run_block(stream, XB, XA, 1024, 8, 16, 60, 30, ei[1], ea[1], 8192,
            blk[1], XG, H, RS, WE, TT);                     // c2 -> (2,1024,16,30)
  run_pool(stream, P12, XA, XB, 512, 1024, 16 * 30);        // -> (2,512,16,30)
  run_block(stream, XB, XA, 512, 16, 32, 30, 15, ei[2], ea[2], 4096,
            blk[2], XG, H, RS, WE, TT);                     // c3 -> (2,512,32,15)
  run_pool(stream, P23, XA, XB, 256, 512, 32 * 15);         // -> (2,256,32,15)
  run_block(stream, XB, XA, 256, 32, 64, 15, 8, ei[3], ea[3], 2048,
            blk[3], XG, H, RS, WE, TT);                     // c4 -> (2,256,64,8)
  run_pool(stream, P34, XA, XB, 128, 256, 64 * 8);          // -> (2,128,64,8)

  // [n,v,c,t] -> [n,c,v,t]
  k_tr_vc<<<cdiv(131072, 256), 256, 0, stream>>>(XB, M1, 128, 64, 8, 131072);

  run_c1x1(stream, fce1W, fce1b, M1, M2, 128, 64, 1);       // (2,128,128,8)
  run_c1x1(stream, fce21W, fce21b, M2, mu, 64, 128, 0);     // mu
  run_c1x1(stream, fce22W, fce22b, M2, logvar, 64, 128, 0); // logvar

  k_reparam<<<cdiv(131072, 256), 256, 0, stream>>>(mu, logvar, eps, ZB, 131072);

  run_c1x1(stream, fcd3W, fcd3b, ZB, M1, 128, 64, 1);       // (2,128,128,8)
  run_c1x1(stream, fcd4W, fcd4b, M1, M2, 64, 128, 1);       // (2,64,128,8)

  // [n,c=64,v=128,t] -> [n,v=128,c=64,t]
  k_tr_vc<<<cdiv(131072, 256), 256, 0, stream>>>(M2, XA, 64, 128, 8, 131072);

  // ---------------- decoder ----------------
  run_pool(stream, P43, XA, XB, 256, 128, 64 * 8);          // -> (2,256,64,8)
  run_block(stream, XB, XA, 256, 64, 32, 8, 15, ei[3], ea[3], 2048,
            blk[4], XG, H, RS, WE, TT);                     // d4 -> (2,256,32,15)
  run_pool(stream, P32, XA, XB, 512, 256, 32 * 15);         // -> (2,512,32,15)
  run_block(stream, XB, XA, 512, 32, 16, 15, 30, ei[2], ea[2], 4096,
            blk[5], XG, H, RS, WE, TT);                     // d3 -> (2,512,16,30)
  run_pool(stream, P21, XA, XB, 1024, 512, 16 * 30);        // -> (2,1024,16,30)
  run_block(stream, XB, XA, 1024, 16, 8, 30, 60, ei[1], ea[1], 8192,
            blk[6], XG, H, RS, WE, TT);                     // d2 -> (2,1024,8,60)
  run_pool(stream, P10, XA, XB, 2048, 1024, 8 * 60);        // -> (2,2048,8,60)
  run_block(stream, XB, recon, 2048, 8, 1, 60, 100, ei[0], ea[0], 16384,
            blk[7], XG, H, RS, WE, TT);                     // d1 -> recon
}
